// KNearestNeighbor_56255481643198
// MI455X (gfx1250) — matrix operand order repaired and checked by reading, not compiled, not run on hardware
//
#include <hip/hip_runtime.h>

typedef __attribute__((ext_vector_type(16))) _Float16 v16h;
typedef __attribute__((ext_vector_type(8)))  _Float16 v8h;
typedef __attribute__((ext_vector_type(8)))  float    v8f;

#define KDIM   128
#define NTRAIN 100000
#define NTEST  1024
#define KNN    5
#define NCLS   10
#define NPART  25           // 6250 N-tiles / 25 partitions
#define TILES_PER_PART 250  // 6250 / 25
#define WAVES  5
#define ITERS  25           // 250 tiles / (5 waves * 2 tiles per wave-iter)

// Workspace layout (bytes); ws base is 256B aligned, offsets are 32B+ aligned.
#define OFF_TH   ((size_t)0)          // train hi f16:  100000*128*2 = 25,600,000
#define OFF_TL   ((size_t)25600000)   // train lo f16:  25,600,000
#define OFF_XH   ((size_t)51200000)   // test hi f16:   1024*128*2 = 262,144
#define OFF_XL   ((size_t)51462144)   // test lo f16:   262,144
#define OFF_TSQ  ((size_t)51724288)   // train sq f32:  100000*4 = 400,000
#define OFF_PD   ((size_t)52124288)   // partial dists: 1024*25*5*4 = 512,000
#define OFF_PI   ((size_t)52636288)   // partial idxs:  512,000
// total: 53,148,288 bytes (~50.7 MiB)

// Sorted-insert into a 5-entry (dist, idx) list; lexicographic (d, idx) order
// reproduces stable top_k tie-breaking (smaller index wins on equal distance).
__device__ __forceinline__ void topk_insert(float d, int id, float bd[KNN], int bi[KNN]) {
  if (d < bd[KNN - 1] || (d == bd[KNN - 1] && id < bi[KNN - 1])) {
    bd[KNN - 1] = d;
    bi[KNN - 1] = id;
#pragma unroll
    for (int j = KNN - 1; j >= 1; --j) {
      bool sw = (bd[j] < bd[j - 1]) || (bd[j] == bd[j - 1] && bi[j] < bi[j - 1]);
      if (sw) {
        float td = bd[j]; bd[j] = bd[j - 1]; bd[j - 1] = td;
        int   ti = bi[j]; bi[j] = bi[j - 1]; bi[j - 1] = ti;
      }
    }
  }
}

// f32 -> (f16 hi, f16 lo) split + row squared norms. One block per row.
__global__ __launch_bounds__(128)
void knn_prep(const float* __restrict__ src, _Float16* __restrict__ dh,
              _Float16* __restrict__ dl, float* __restrict__ sq) {
  __shared__ float red[128];
  const int row = blockIdx.x;
  const int t = threadIdx.x;
  const size_t idx = (size_t)row * KDIM + t;
  const float v = src[idx];
  const _Float16 h = (_Float16)v;
  dh[idx] = h;
  dl[idx] = (_Float16)(v - (float)h);
  if (sq != nullptr) {
    red[t] = v * v;
    __syncthreads();
#pragma unroll
    for (int s = 64; s > 0; s >>= 1) {
      if (t < s) red[t] += red[t + s];
      __syncthreads();
    }
    if (t == 0) sq[row] = red[0];
  }
}

// GEMM (WMMA f16 hi/lo split) fused with per-row top-5 over one N-partition.
// Grid: x = partition (25), y = test-row tile (64). Block: 160 threads = 5
// waves; each wave independently processes 2 N-tiles per iteration with two
// accumulators (2-way WMMA ILP). No block barriers in the hot loop: LDS tiles
// are per-wave and CDNA5 DS ops from one wave complete in order (DScnt).
__global__ __launch_bounds__(160)
void knn_gemm_topk(const _Float16* __restrict__ Xh, const _Float16* __restrict__ Xl,
                   const _Float16* __restrict__ Th, const _Float16* __restrict__ Tl,
                   const float* __restrict__ Tsq,
                   float* __restrict__ PD, int* __restrict__ PI) {
  __shared__ float distT[WAVES][2][16][17];  // row padded to 17: conflict-free reads
  __shared__ float mD[WAVES][2][16][KNN];
  __shared__ int   mI[WAVES][2][16][KNN];

  const int tid   = threadIdx.x;
  const int wave  = tid >> 5;
  const int lane  = tid & 31;
  const int part  = blockIdx.x;
  const int mBase = blockIdx.y * 16;
  const int col   = lane & 15;

  // ---- A fragments (16 test rows x 128 K, hi+lo) live in VGPRs all loop.
  // A 16x32 layout (matches B): lanes 0-15 hold K 0..15 of row=lane,
  // lanes 16-31 hold K 16..31 of row=lane-16.
  const int aRow = mBase + col;
  const _Float16* __restrict__ ah = Xh + (size_t)aRow * KDIM;
  const _Float16* __restrict__ al = Xl + (size_t)aRow * KDIM;
  const int aOff = (lane < 16) ? 0 : 16;
  v16h Ah[4], Al[4];
#pragma unroll
  for (int kf = 0; kf < 4; ++kf) {
    v8h h0 = *(const v8h*)(ah + 32 * kf + aOff);
    v8h h1 = *(const v8h*)(ah + 32 * kf + 8 + aOff);
    v8h l0 = *(const v8h*)(al + 32 * kf + aOff);
    v8h l1 = *(const v8h*)(al + 32 * kf + 8 + aOff);
#pragma unroll
    for (int i = 0; i < 8; ++i) {
      Ah[kf][i] = h0[i]; Ah[kf][8 + i] = h1[i];
      Al[kf][i] = l0[i]; Al[kf][8 + i] = l1[i];
    }
  }

  float bd[KNN]; int bi[KNN];
#pragma unroll
  for (int j = 0; j < KNN; ++j) { bd[j] = __builtin_huge_valf(); bi[j] = 0x7fffffff; }

  // B 32x16 layout: lanes 0-15 hold K 0..15 of col=lane, lanes 16-31 K 16..31.
  const int bOff = (lane < 16) ? 0 : 16;
  const int myHalf = lane >> 4;  // selection: lanes 0-15 scan tile0, 16-31 tile1

  for (int it = 0; it < ITERS; ++it) {
    const int tile0 = part * TILES_PER_PART + (it * WAVES + wave) * 2;
    const int nB0 = tile0 * 16;
    const int nB1 = nB0 + 16;
    const _Float16* __restrict__ bh0 = Th + (size_t)(nB0 + col) * KDIM + bOff;
    const _Float16* __restrict__ bl0 = Tl + (size_t)(nB0 + col) * KDIM + bOff;
    const _Float16* __restrict__ bh1 = Th + (size_t)(nB1 + col) * KDIM + bOff;
    const _Float16* __restrict__ bl1 = Tl + (size_t)(nB1 + col) * KDIM + bOff;

    v8f c0 = {}, c1 = {};
#pragma unroll
    for (int kf = 0; kf < 4; ++kf) {
      v16h Bh0 = *(const v16h*)(bh0 + 32 * kf);
      v16h Bh1 = *(const v16h*)(bh1 + 32 * kf);
      v16h Bl0 = *(const v16h*)(bl0 + 32 * kf);
      v16h Bl1 = *(const v16h*)(bl1 + 32 * kf);
      // cross ~= hi*hi + hi*lo + lo*hi (near-f32 accuracy); c0/c1 independent
      c0 = __builtin_amdgcn_wmma_f32_16x16x32_f16(false, Ah[kf], false, Bh0, (short)0, c0, false, false);
      c1 = __builtin_amdgcn_wmma_f32_16x16x32_f16(false, Ah[kf], false, Bh1, (short)0, c1, false, false);
      c0 = __builtin_amdgcn_wmma_f32_16x16x32_f16(false, Ah[kf], false, Bl0, (short)0, c0, false, false);
      c1 = __builtin_amdgcn_wmma_f32_16x16x32_f16(false, Ah[kf], false, Bl1, (short)0, c1, false, false);
      c0 = __builtin_amdgcn_wmma_f32_16x16x32_f16(false, Al[kf], false, Bh0, (short)0, c0, false, false);
      c1 = __builtin_amdgcn_wmma_f32_16x16x32_f16(false, Al[kf], false, Bh1, (short)0, c1, false, false);
    }

    // Ranking key: ||t||^2 - 2*(x.t)  (||x||^2 and sqrt are rank-invariant).
    const float tsq0 = Tsq[nB0 + col];
    const float tsq1 = Tsq[nB1 + col];
    const int rowHalf = myHalf * 8;  // C/D layout: VGPR r -> M = r + 8*(lane>=16)
#pragma unroll
    for (int r = 0; r < 8; ++r) {
      distT[wave][0][rowHalf + r][col] = tsq0 - 2.0f * c0[r];
      distT[wave][1][rowHalf + r][col] = tsq1 - 2.0f * c1[r];
    }

    // Wave-local visibility: same-wave DS ops complete in order; wait + block
    // compiler reordering. No cross-wave traffic on this LDS region.
    asm volatile("s_wait_dscnt 0" ::: "memory");

    const int myBase = myHalf ? nB1 : nB0;
#pragma unroll
    for (int cc = 0; cc < 16; ++cc)
      topk_insert(distT[wave][myHalf][col][cc], myBase + cc, bd, bi);

    asm volatile("" ::: "memory");  // keep next iter's stores after these reads
  }

  // ---- merge 10 per-wave lists per row -> per-partition top-5
#pragma unroll
  for (int j = 0; j < KNN; ++j) {
    mD[wave][myHalf][col][j] = bd[j];
    mI[wave][myHalf][col][j] = bi[j];
  }
  __syncthreads();

  if (tid < 16) {
    float fd[KNN]; int fi[KNN];
#pragma unroll
    for (int j = 0; j < KNN; ++j) { fd[j] = __builtin_huge_valf(); fi[j] = 0x7fffffff; }
    for (int w = 0; w < WAVES; ++w)
      for (int h = 0; h < 2; ++h)
#pragma unroll
        for (int j = 0; j < KNN; ++j)
          topk_insert(mD[w][h][tid][j], mI[w][h][tid][j], fd, fi);
    const int row = mBase + tid;
    float* pd = PD + ((size_t)row * NPART + part) * KNN;
    int*   pi = PI + ((size_t)row * NPART + part) * KNN;
#pragma unroll
    for (int j = 0; j < KNN; ++j) { pd[j] = fd[j]; pi[j] = fi[j]; }
  }
}

// Merge 25 partition lists per row, gather labels, majority vote (smallest
// label wins ties), emit f32 prediction.
__global__ __launch_bounds__(256)
void knn_vote(const float* __restrict__ PD, const int* __restrict__ PI,
              const int* __restrict__ y, float* __restrict__ out) {
  const int row = blockIdx.x * blockDim.x + threadIdx.x;
  if (row >= NTEST) return;

  float bd[KNN]; int bi[KNN];
#pragma unroll
  for (int j = 0; j < KNN; ++j) { bd[j] = __builtin_huge_valf(); bi[j] = 0x7fffffff; }

  const float* pd = PD + (size_t)row * NPART * KNN;
  const int*   pi = PI + (size_t)row * NPART * KNN;
  for (int j = 0; j < NPART * KNN; ++j)
    topk_insert(pd[j], pi[j], bd, bi);

  int cnt[NCLS];
#pragma unroll
  for (int cls = 0; cls < NCLS; ++cls) cnt[cls] = 0;
#pragma unroll
  for (int j = 0; j < KNN; ++j) {
    const int lab = y[bi[j]];
#pragma unroll
    for (int cls = 0; cls < NCLS; ++cls) cnt[cls] += (lab == cls) ? 1 : 0;
  }
  int best = 0, bc = cnt[0];
#pragma unroll
  for (int cls = 1; cls < NCLS; ++cls)
    if (cnt[cls] > bc) { bc = cnt[cls]; best = cls; }
  out[row] = (float)best;
}

extern "C" void kernel_launch(void* const* d_in, const int* in_sizes, int n_in,
                              void* d_out, int out_size, void* d_ws, size_t ws_size,
                              hipStream_t stream) {
  (void)in_sizes; (void)n_in; (void)out_size; (void)ws_size;
  const float* X  = (const float*)d_in[0];   // [1024, 128]
  const float* Xt = (const float*)d_in[1];   // [100000, 128]
  const int*   y  = (const int*)d_in[2];     // [100000]
  // d_in[3] is k == 5 (hardcoded per setup_inputs)

  char* ws = (char*)d_ws;
  _Float16* Th  = (_Float16*)(ws + OFF_TH);
  _Float16* Tl  = (_Float16*)(ws + OFF_TL);
  _Float16* Xh  = (_Float16*)(ws + OFF_XH);
  _Float16* Xl  = (_Float16*)(ws + OFF_XL);
  float*    Tsq = (float*)(ws + OFF_TSQ);
  float*    PD  = (float*)(ws + OFF_PD);
  int*      PI  = (int*)(ws + OFF_PI);

  knn_prep<<<NTRAIN, 128, 0, stream>>>(Xt, Th, Tl, Tsq);
  knn_prep<<<NTEST, 128, 0, stream>>>(X, Xh, Xl, nullptr);

  dim3 grid(NPART, NTEST / 16);
  knn_gemm_topk<<<grid, WAVES * 32, 0, stream>>>(Xh, Xl, Th, Tl, Tsq, PD, PI);

  knn_vote<<<(NTEST + 255) / 256, 256, 0, stream>>>(PD, PI, y, (float*)d_out);
}
